// VectorQuantizer_81905026335345
// MI455X (gfx1250) — matrix operand order repaired and checked by reading, not compiled, not run on hardware
//
#include <hip/hip_runtime.h>

// CDNA5 / gfx1250 vector-quantizer: fused bf16-split WMMA GEMM + row argmin.
typedef __attribute__((ext_vector_type(16))) __bf16 v16bf;
typedef __attribute__((ext_vector_type(8)))  float  v8f;

#define NUM_EMB   1024
#define EMB_DIM   32
#define HW        4096          // 64*64
#define NCHAN     32
#define ROWS_WG   256           // rows (b,h,w) handled per workgroup
#define N_TOTAL   (32 * HW)     // 131072 query rows
#define ZQ_ELEMS  (32 * NCHAN * HW)  // 4,194,304

// dynamic LDS layout
#define SM_HI   0                       // 1024*32 bf16 = 65536 B
#define SM_LO   65536                   // 1024*32 bf16 = 65536 B
#define SM_EB   131072                  // 1024 f32     = 4096 B
#define SM_IDX  135168                  // 256 int      = 1024 B
#define SM_TOT  136192

__global__ __launch_bounds__(256) void vq_wmma_kernel(
    const float* __restrict__ z,     // [32][32][64][64]
    const float* __restrict__ emb,   // [1024][32]
    float* __restrict__ out)         // z_q (4194304) ++ idx (131072)
{
  extern __shared__ char smem[];
  __bf16* sHi = (__bf16*)(smem + SM_HI);
  __bf16* sLo = (__bf16*)(smem + SM_LO);
  float*  sEb = (float*)(smem + SM_EB);
  int*    sIdx = (int*)(smem + SM_IDX);

  const int t = threadIdx.x;

  // ---- Phase 1: stage codebook into LDS as split bf16 (hi+lo), fold -0.5*||e||^2
  #pragma unroll
  for (int cb = 0; cb < NUM_EMB / 256; ++cb) {
    const int code = cb * 256 + t;
    const float* er = emb + code * EMB_DIM;
    float ss = 0.f;
    #pragma unroll
    for (int j = 0; j < EMB_DIM; ++j) {
      const float x = er[j];
      ss += x * x;
      const __bf16 h = (__bf16)x;
      const __bf16 l = (__bf16)(x - (float)h);
      sHi[code * EMB_DIM + j] = h;
      sLo[code * EMB_DIM + j] = l;
    }
    sEb[code] = -0.5f * ss;
  }
  __syncthreads();

  // ---- Phase 2: per-wave, two 16-row tiles vs all 1024 codes
  const int wave = t >> 5;
  const int lane = t & 31;
  const int col  = lane & 15;     // N position within 16-wide tile
  const int half = lane >> 4;     // lane group (affects K layout)

  const int nBase  = blockIdx.x * ROWS_WG;          // 256-aligned -> single batch image
  const int bIdx   = nBase >> 12;                   // 4096 (h,w) per image
  const int hwBase = (nBase & (HW - 1)) + wave * 32;
  const float* zb  = z + (size_t)bIdx * (NCHAN * HW);

  // A fragments (contiguous K order, same as B): element e -> K = 16*half+e
  v16bf a0h, a0l, a1h, a1l;
  #pragma unroll
  for (int e = 0; e < 16; ++e) {
    const int c = 16 * half + e;
    const float x0 = zb[c * HW + hwBase + col];
    const float x1 = zb[c * HW + hwBase + 16 + col];
    const __bf16 h0 = (__bf16)x0;
    const __bf16 h1 = (__bf16)x1;
    a0h[e] = h0;  a0l[e] = (__bf16)(x0 - (float)h0);
    a1h[e] = h1;  a1l[e] = (__bf16)(x1 - (float)h1);
  }

  float bv0[8], bv1[8];
  int   bi0[8], bi1[8];
  #pragma unroll
  for (int r = 0; r < 8; ++r) {
    bv0[r] = -3.4e38f; bv1[r] = -3.4e38f; bi0[r] = 0; bi1[r] = 0;
  }

  #pragma unroll 4
  for (int tt = 0; tt < NUM_EMB / 16; ++tt) {
    const int code = tt * 16 + col;
    // B fragment (32x16 bf16): lane holds contiguous K in [16*half, 16*half+16)
    const __bf16* ph = sHi + code * EMB_DIM + 16 * half;
    const __bf16* pl = sLo + code * EMB_DIM + 16 * half;
    union { v16bf v; uint4 q[2]; } bh, bl;
    bh.q[0] = ((const uint4*)ph)[0];  bh.q[1] = ((const uint4*)ph)[1];
    bl.q[0] = ((const uint4*)pl)[0];  bl.q[1] = ((const uint4*)pl)[1];
    const float ebv = sEb[code];      // -0.5*||e||^2, broadcast per column (matches C layout)

    v8f acc0 = {ebv, ebv, ebv, ebv, ebv, ebv, ebv, ebv};
    v8f acc1 = acc0;
    // score = z.e - 0.5*||e||^2  via split bf16: hi*hi + hi*lo + lo*hi
    acc0 = __builtin_amdgcn_wmma_f32_16x16x32_bf16(false, a0h, false, bh.v, (short)0, acc0, false, false);
    acc1 = __builtin_amdgcn_wmma_f32_16x16x32_bf16(false, a1h, false, bh.v, (short)0, acc1, false, false);
    acc0 = __builtin_amdgcn_wmma_f32_16x16x32_bf16(false, a0h, false, bl.v, (short)0, acc0, false, false);
    acc1 = __builtin_amdgcn_wmma_f32_16x16x32_bf16(false, a1h, false, bl.v, (short)0, acc1, false, false);
    acc0 = __builtin_amdgcn_wmma_f32_16x16x32_bf16(false, a0l, false, bh.v, (short)0, acc0, false, false);
    acc1 = __builtin_amdgcn_wmma_f32_16x16x32_bf16(false, a1l, false, bh.v, (short)0, acc1, false, false);

    // running argmax (== argmin of distance); strict '>' keeps lowest index on ties
    #pragma unroll
    for (int r = 0; r < 8; ++r) {
      if (acc0[r] > bv0[r]) { bv0[r] = acc0[r]; bi0[r] = code; }
      if (acc1[r] > bv1[r]) { bv1[r] = acc1[r]; bi1[r] = code; }
    }
  }

  // ---- Cross-lane argmax reduction over the 16 N-lanes (each 16-lane half independent)
  #pragma unroll
  for (int m = 1; m < 16; m <<= 1) {
    #pragma unroll
    for (int r = 0; r < 8; ++r) {
      float ov = __shfl_xor(bv0[r], m, 32);
      int   oi = __shfl_xor(bi0[r], m, 32);
      if (ov > bv0[r] || (ov == bv0[r] && oi < bi0[r])) { bv0[r] = ov; bi0[r] = oi; }
      ov = __shfl_xor(bv1[r], m, 32);
      oi = __shfl_xor(bi1[r], m, 32);
      if (ov > bv1[r] || (ov == bv1[r] && oi < bi1[r])) { bv1[r] = ov; bi1[r] = oi; }
    }
  }

  // C/D layout: lanes 0-15 reg r -> row r; lanes 16-31 reg r -> row 8+r
  if (col == 0) {
    const int base = wave * 32 + half * 8;
    #pragma unroll
    for (int r = 0; r < 8; ++r) sIdx[base + r] = bi0[r];
    #pragma unroll
    for (int r = 0; r < 8; ++r) sIdx[base + 16 + r] = bi1[r];
  }
  __syncthreads();

  // ---- Phase 3: outputs. idx (as float) + z_q gathered from global emb (exact fp32),
  // channel-major stores: 256 consecutive floats per channel -> fully coalesced.
  const int myIdx = sIdx[t];
  out[ZQ_ELEMS + nBase + t] = (float)myIdx;
  const float* eRow = emb + myIdx * EMB_DIM;
  float* ob = out + (size_t)bIdx * (NCHAN * HW) + (nBase & (HW - 1)) + t;
  #pragma unroll
  for (int c = 0; c < NCHAN; ++c)
    ob[c * HW] = eRow[c];
}

extern "C" void kernel_launch(void* const* d_in, const int* in_sizes, int n_in,
                              void* d_out, int out_size, void* d_ws, size_t ws_size,
                              hipStream_t stream) {
  (void)in_sizes; (void)n_in; (void)out_size; (void)d_ws; (void)ws_size;
  const float* z   = (const float*)d_in[0];
  const float* emb = (const float*)d_in[1];
  float* out = (float*)d_out;

  static const int smem = SM_TOT;  // 133 KB dynamic LDS (fits 320 KB/WGP)
  hipFuncSetAttribute((const void*)vq_wmma_kernel,
                      hipFuncAttributeMaxDynamicSharedMemorySize, smem);
  vq_wmma_kernel<<<N_TOTAL / ROWS_WG, 256, smem, stream>>>(z, emb, out);
}
